// CrossAttentionBlock_34918084116968
// MI455X (gfx1250) — matrix operand order repaired and checked by reading, not compiled, not run on hardware
//
#include <hip/hip_runtime.h>
#include <stdint.h>

typedef unsigned short u16;
typedef __attribute__((ext_vector_type(16))) __bf16 v16bf;
typedef __attribute__((ext_vector_type(8)))  float  v8f;

union frag_u { v16bf v; uint4 u[2]; };

#define N_SENT 16384
#define N_FUNC 8192
#define DIM    256
#define NKB    (N_FUNC / 32)

static __device__ __forceinline__ u16 f2bf(float f) {
  union { float f; uint32_t u; } a; a.f = f;
  uint32_t r = a.u + 0x7FFFu + ((a.u >> 16) & 1u);   // round-to-nearest-even
  return (u16)(r >> 16);
}

// ---- CDNA5 async global->LDS DMA (ASYNCcnt-tracked), via inline asm ----
// global_load_async_to_lds_b128: vdst = LDS byte offset (addr[31:0] of the
// flat pointer = wave-relative LDS address), vaddr = 64-bit global address.
static __device__ __forceinline__ void copy16_async(const u16* g, u16* l) {
  uint32_t loff = (uint32_t)(uintptr_t)l;   // low 32 bits of flat LDS ptr = LDS offset
  asm volatile("global_load_async_to_lds_b128 %0, %1, off"
               :: "v"(loff), "v"(g)
               : "memory");
}

static __device__ __forceinline__ void wait_async_all() {
  asm volatile("s_wait_asynccnt 0x0" ::: "memory");
}

// ---------------- prep: fp32 -> bf16 (optionally scaled) ----------------
__global__ void cvt_bf16_kernel(const float* __restrict__ x, u16* __restrict__ y,
                                float scale, int n4) {
  int i = blockIdx.x * blockDim.x + threadIdx.x;
  if (i >= n4) return;
  float4 v = ((const float4*)x)[i];
  uint2 o;
  o.x = (uint32_t)f2bf(v.x * scale) | ((uint32_t)f2bf(v.y * scale) << 16);
  o.y = (uint32_t)f2bf(v.z * scale) | ((uint32_t)f2bf(v.w * scale) << 16);
  ((uint2*)y)[i] = o;
}

// ------- prep: V = func_embed @ Wv_w^T + b, stored transposed bf16 -------
__global__ __launch_bounds__(256) void gemm_v_kernel(const float* __restrict__ func,
                                                     const float* __restrict__ W,
                                                     const float* __restrict__ b,
                                                     u16* __restrict__ Vt) {
  int id = blockIdx.x * 256 + threadIdx.x;   // over N_FUNC*DIM
  int n = id >> 8;
  int d = id & 255;
  const float4* fr = (const float4*)(func + (size_t)n * DIM);
  const float4* wr = (const float4*)(W    + (size_t)d * DIM);
  float acc = b[d];
  #pragma unroll 8
  for (int k = 0; k < DIM / 4; ++k) {
    float4 a = fr[k], w = wr[k];
    acc += a.x * w.x + a.y * w.y + a.z * w.z + a.w * w.w;
  }
  Vt[(size_t)d * N_FUNC + n] = f2bf(acc);    // [DIM][N_FUNC]
}

// ---------------- fused flash-attention, double-buffered async tiles ----------------
// grid: N_SENT/128 blocks, 256 threads (8 waves). Wave owns 16 query rows.
__global__ __launch_bounds__(256) void fa_kernel(const u16* __restrict__ Qb,
                                                 const u16* __restrict__ Kb,
                                                 const u16* __restrict__ Vt,
                                                 float* __restrict__ out) {
  __shared__ alignas(16) u16 sK[2][32 * DIM];     // 2 x 16 KB
  __shared__ alignas(16) u16 sV[2][DIM * 32];     // 2 x 16 KB
  __shared__ alignas(16) u16 sP[8 * 16 * 32];     // 8 KB per-wave P scratch

  const int tid  = threadIdx.x;
  const int wave = tid >> 5;
  const int lane = tid & 31;
  const int lh   = lane >> 4;
  const int ll   = lane & 15;
  const int q0   = (blockIdx.x * 8 + wave) * 16;

  // async tile fetch: each thread moves 64B of K and 64B of Vt
  const int krow = tid >> 3, kch = tid & 7;
  auto issue_tile = [&](int kb, int buf) {
    const u16* gk = Kb + (size_t)(kb * 32 + krow) * DIM + kch * 32;
    u16* lk = &sK[buf][krow * DIM + kch * 32];
    #pragma unroll
    for (int i = 0; i < 4; ++i) copy16_async(gk + i * 8, lk + i * 8);
    const u16* gv = Vt + (size_t)tid * N_FUNC + kb * 32;
    u16* lv = &sV[buf][tid * 32];
    #pragma unroll
    for (int i = 0; i < 4; ++i) copy16_async(gv + i * 8, lv + i * 8);
  };

  // -------- Q fragments in registers: 8 chunks of 16x32 (A layout) --------
  v16bf qf[8];
  {
    const u16* qrow = Qb + (size_t)(q0 + ll) * DIM;
    #pragma unroll
    for (int c = 0; c < 8; ++c) {
      frag_u f;
      f.u[0] = *(const uint4*)(qrow + c * 32 + lh * 8);
      f.u[1] = *(const uint4*)(qrow + c * 32 + 16 + lh * 8);
      qf[c] = f.v;
    }
  }

  const v8f vzero = {0.f, 0.f, 0.f, 0.f, 0.f, 0.f, 0.f, 0.f};
  v8f O[16];
  #pragma unroll
  for (int t = 0; t < 16; ++t) O[t] = vzero;
  float m[8], lsum[8];
  #pragma unroll
  for (int r = 0; r < 8; ++r) { m[r] = -INFINITY; lsum[r] = 0.f; }

  issue_tile(0, 0);

  for (int kb = 0; kb < NKB; ++kb) {
    wait_async_all();        // my share of tile kb is in LDS
    __syncthreads();         // everyone's tile kb present; compute(kb-1) done by all
    if (kb + 1 < NKB) issue_tile(kb + 1, (kb + 1) & 1);   // DMA overlaps compute
    const u16* curK = sK[kb & 1];
    const u16* curV = sV[kb & 1];

    // ---- S = (Q/16) @ K^T : two 16x16 tiles over K-dim 256 ----
    v8f s0 = vzero, s1 = vzero;
    #pragma unroll
    for (int c = 0; c < 8; ++c) {
      frag_u b0, b1;
      const u16* kp0 = curK + ll * DIM + c * 32 + lh * 8;    // keys 0..15
      const u16* kp1 = kp0 + 16 * DIM;                       // keys 16..31
      b0.u[0] = *(const uint4*)(kp0);  b0.u[1] = *(const uint4*)(kp0 + 16);
      b1.u[0] = *(const uint4*)(kp1);  b1.u[1] = *(const uint4*)(kp1 + 16);
      s0 = __builtin_amdgcn_wmma_f32_16x16x32_bf16(false, qf[c], false, b0.v,
                                                   (short)0, s0, false, false);
      s1 = __builtin_amdgcn_wmma_f32_16x16x32_bf16(false, qf[c], false, b1.v,
                                                   (short)0, s1, false, false);
    }

    // ---- online softmax (row stats across 16-lane halves) ----
    float al[8];
    #pragma unroll
    for (int r = 0; r < 8; ++r) {
      float t = fmaxf(s0[r], s1[r]);
      t = fmaxf(t, __shfl_xor(t, 1, 32));
      t = fmaxf(t, __shfl_xor(t, 2, 32));
      t = fmaxf(t, __shfl_xor(t, 4, 32));
      t = fmaxf(t, __shfl_xor(t, 8, 32));
      float mn = fmaxf(m[r], t);
      al[r] = __expf(m[r] - mn);
      m[r] = mn;
    }
    #pragma unroll
    for (int r = 0; r < 8; ++r) {
      float p0 = __expf(s0[r] - m[r]);
      float p1 = __expf(s1[r] - m[r]);
      s0[r] = p0; s1[r] = p1;
      float t = p0 + p1;
      t += __shfl_xor(t, 1, 32);
      t += __shfl_xor(t, 2, 32);
      t += __shfl_xor(t, 4, 32);
      t += __shfl_xor(t, 8, 32);
      lsum[r] = lsum[r] * al[r] + t;
    }

    // ---- relayout P: C layout -> A layout via per-wave LDS scratch ----
    {
      u16* pw = sP + wave * 512;
      #pragma unroll
      for (int r = 0; r < 8; ++r) {
        int row = r + 8 * lh;
        pw[row * 32 + ll]      = f2bf(s0[r]);
        pw[row * 32 + 16 + ll] = f2bf(s1[r]);
      }
    }
    frag_u pf;
    {
      const u16* pr = sP + wave * 512 + ll * 32 + lh * 8;
      pf.u[0] = *(const uint4*)(pr);
      pf.u[1] = *(const uint4*)(pr + 16);
    }

    // ---- rescale accumulator, then O += P @ V ----
    #pragma unroll
    for (int t = 0; t < 16; ++t) {
      #pragma unroll
      for (int r = 0; r < 8; ++r) O[t][r] *= al[r];
    }
    #pragma unroll
    for (int t = 0; t < 16; ++t) {
      frag_u bv;
      const u16* vp = curV + (t * 16 + ll) * 32 + lh * 8;
      bv.u[0] = *(const uint4*)(vp);  bv.u[1] = *(const uint4*)(vp + 16);
      O[t] = __builtin_amdgcn_wmma_f32_16x16x32_bf16(false, pf.v, false, bv.v,
                                                     (short)0, O[t], false, false);
    }
  }

  // ---- finalize: O /= l, store fp32 ----
  float inv[8];
  #pragma unroll
  for (int r = 0; r < 8; ++r) inv[r] = 1.0f / lsum[r];
  #pragma unroll
  for (int t = 0; t < 16; ++t) {
    #pragma unroll
    for (int r = 0; r < 8; ++r) {
      out[(size_t)(q0 + r + 8 * lh) * DIM + t * 16 + ll] = O[t][r] * inv[r];
    }
  }
}

extern "C" void kernel_launch(void* const* d_in, const int* in_sizes, int n_in,
                              void* d_out, int out_size, void* d_ws, size_t ws_size,
                              hipStream_t stream) {
  const float* func = (const float*)d_in[0];   // [8192, 256]
  const float* sent = (const float*)d_in[1];   // [16384, 256]
  const float* Wv_w = (const float*)d_in[2];   // [256, 256]
  const float* Wv_b = (const float*)d_in[3];   // [256]
  float* out = (float*)d_out;                  // [16384, 256]

  u16* Qb = (u16*)d_ws;                        // 8 MB
  u16* Kb = Qb + (size_t)N_SENT * DIM;         // 4 MB
  u16* Vt = Kb + (size_t)N_FUNC * DIM;         // 4 MB (transposed)

  // Q pre-scaled by 1/sqrt(256) = 1/16 (exact power of two in bf16)
  cvt_bf16_kernel<<<(N_SENT * DIM / 4 + 255) / 256, 256, 0, stream>>>(sent, Qb, 0.0625f, N_SENT * DIM / 4);
  cvt_bf16_kernel<<<(N_FUNC * DIM / 4 + 255) / 256, 256, 0, stream>>>(func, Kb, 1.0f, N_FUNC * DIM / 4);
  gemm_v_kernel<<<N_FUNC, 256, 0, stream>>>(func, Wv_w, Wv_b, Vt);
  fa_kernel<<<N_SENT / 128, 256, 0, stream>>>(Qb, Kb, Vt, out);
}
